// Sequence_1022202217223
// MI455X (gfx1250) — matrix operand order repaired and checked by reading, not compiled, not run on hardware
//
#include <hip/hip_runtime.h>
#include <math.h>

// ---------------------------------------------------------------------------
// 2-layer LSTM (T=2048, B=128, IN=64, H=256) + linear head, persistent kernel.
// 16 workgroups x 256 threads. WG w owns hidden slice [16w, 16w+16).
// Weights live in LDS in WMMA-fragment order (f16). h state ping-pongs in
// f16 global workspace; c state is f32 workspace, thread-private.
// Matrix math: v_wmma_f32_16x16x32_f16 (A=f16 16x32, B=f16 32x16, C=f32).
// Gate nonlinearities use v_exp_f32 + v_rcp_f32 to keep the serial
// recurrence tail (GEMM -> gates -> grid barrier) short.
// ---------------------------------------------------------------------------

#define T_STEPS 2048
#define BATCH   128
#define HID     256
#define NWG     16
#define HB      (BATCH * HID)     // elements of one h buffer
#define NFRAG   104               // 8 (ih0) + 32 (hh0) + 32 (ih1) + 32 (hh1)

typedef _Float16 half8  __attribute__((ext_vector_type(8)));
typedef _Float16 half16 __attribute__((ext_vector_type(16)));
typedef float    v8f    __attribute__((ext_vector_type(8)));

// ---------------- fragment loads ----------------

// B fragment (32x16, f16): lane L holds N column L%16, halves = K in AMD table order.
// LDS image per fragment: 1024B; lane L: bytes [L*16] = halves 0..7, [512+L*16] = halves 8..15.
__device__ inline half16 load_b_frag(const _Float16* wf, int f, int lane) {
  const _Float16* p = wf + f * 512 + lane * 8;
  half8 lo = *(const half8*)(p);
  half8 hi = *(const half8*)(p + 256);
  return __builtin_shufflevector(lo, hi, 0,1,2,3,4,5,6,7,8,9,10,11,12,13,14,15);
}

// A fragment (16x32, f16) from an f16 [.,256] buffer.
// lanes 0-15: row=m, halves 0-7 = K c0..c0+7, halves 8-15 = K c0+16..c0+23 (c0 = kc)
// lanes 16-31: same rows, c0 = kc+8.
__device__ inline half16 load_h_frag(const _Float16* hbuf, int row, int c0) {
  const _Float16* p = hbuf + row * HID + c0;
  half8 lo = *(const half8*)(p);
  half8 hi = *(const half8*)(p + 16);
  return __builtin_shufflevector(lo, hi, 0,1,2,3,4,5,6,7,8,9,10,11,12,13,14,15);
}

// A fragment from f32 x_t [128,64] with inline f32->f16 conversion.
__device__ inline half16 load_x_frag(const float* xt, int row, int c0) {
  const float* p = xt + row * 64 + c0;
  float4 a = *(const float4*)(p);
  float4 b = *(const float4*)(p + 4);
  float4 c = *(const float4*)(p + 16);
  float4 d = *(const float4*)(p + 20);
  half16 r;
  r[0]=(_Float16)a.x; r[1]=(_Float16)a.y; r[2]=(_Float16)a.z; r[3]=(_Float16)a.w;
  r[4]=(_Float16)b.x; r[5]=(_Float16)b.y; r[6]=(_Float16)b.z; r[7]=(_Float16)b.w;
  r[8]=(_Float16)c.x; r[9]=(_Float16)c.y; r[10]=(_Float16)c.z; r[11]=(_Float16)c.w;
  r[12]=(_Float16)d.x; r[13]=(_Float16)d.y; r[14]=(_Float16)d.z; r[15]=(_Float16)d.w;
  return r;
}

// one A against the 4 gate tiles (i,f,g,o) of a K-chunk
__device__ inline void wmma4(half16 A, const _Float16* wf, int fb, int lane,
                             v8f& zi, v8f& zf, v8f& zg, v8f& zo) {
  zi = __builtin_amdgcn_wmma_f32_16x16x32_f16(false, A, false, load_b_frag(wf, fb+0, lane), (short)0, zi, false, false);
  zf = __builtin_amdgcn_wmma_f32_16x16x32_f16(false, A, false, load_b_frag(wf, fb+1, lane), (short)0, zf, false, false);
  zg = __builtin_amdgcn_wmma_f32_16x16x32_f16(false, A, false, load_b_frag(wf, fb+2, lane), (short)0, zg, false, false);
  zo = __builtin_amdgcn_wmma_f32_16x16x32_f16(false, A, false, load_b_frag(wf, fb+3, lane), (short)0, zo, false, false);
}

// fast nonlinearities on hardware transcendentals (v_exp_f32 / v_rcp_f32);
// correct saturation at +/-inf, error ~1ulp of f32 << f16 state quantization.
__device__ inline float sigmoidf_(float v) {
  return __builtin_amdgcn_rcpf(1.0f + __expf(-v));
}
__device__ inline float tanhf_(float v) {
  return 2.0f * __builtin_amdgcn_rcpf(1.0f + __expf(-2.0f * v)) - 1.0f;
}

// gate nonlinearity + c/h update. C/D tile layout: VGPR r, lanes 0-15: M=r,
// N=lane; lanes 16-31: M=8+r, N=lane-16.
__device__ inline void gate_update(const v8f& zi, const v8f& zf, const v8f& zg, const v8f& zo,
                                   float bi, float bf, float bg, float bo,
                                   float* cst, _Float16* hout, int mbase, int hidx) {
#pragma unroll
  for (int r = 0; r < 8; ++r) {
    int b = mbase + r;
    float iv = sigmoidf_(zi[r] + bi);
    float fv = sigmoidf_(zf[r] + bf);
    float gv = tanhf_(zg[r] + bg);
    float ov = sigmoidf_(zo[r] + bo);
    float cn = fv * cst[b * HID + hidx] + iv * gv;
    cst[b * HID + hidx] = cn;
    hout[b * HID + hidx] = (_Float16)(ov * tanhf_(cn));
  }
}

// software grid barrier over NWG persistent workgroups (monotonic epoch)
__device__ inline void grid_barrier(unsigned* cnt, unsigned target) {
  __builtin_amdgcn_fence(__ATOMIC_RELEASE, "agent");
  __syncthreads();
  if (threadIdx.x == 0) {
    __hip_atomic_fetch_add(cnt, 1u, __ATOMIC_RELAXED, __HIP_MEMORY_SCOPE_AGENT);
    while (__hip_atomic_load(cnt, __ATOMIC_RELAXED, __HIP_MEMORY_SCOPE_AGENT) < target) {
      __builtin_amdgcn_s_sleep(2);
    }
  }
  __syncthreads();
  __builtin_amdgcn_fence(__ATOMIC_ACQUIRE, "agent");
}

// ---------------- prep: reset barrier counter, seed h/c state ----------------
__global__ void lstm_prep(const float* __restrict__ h0, const float* __restrict__ c0,
                          _Float16* __restrict__ h0buf, _Float16* __restrict__ h1buf,
                          float* __restrict__ c0ws, float* __restrict__ c1ws,
                          unsigned* __restrict__ sync) {
  int i = blockIdx.x * blockDim.x + threadIdx.x;
  if (i == 0) sync[0] = 0u;
  if (i < HB) {
    h0buf[i] = (_Float16)h0[i];
    h1buf[i] = (_Float16)h0[HB + i];
    c0ws[i] = c0[i];
    c1ws[i] = c0[HB + i];
  }
}

// ---------------- persistent LSTM kernel ----------------
__global__ __launch_bounds__(256) void lstm_persistent(
    const float* __restrict__ x,
    const float* __restrict__ Wih0, const float* __restrict__ Whh0,
    const float* __restrict__ bih0, const float* __restrict__ bhh0,
    const float* __restrict__ Wih1, const float* __restrict__ Whh1,
    const float* __restrict__ bih1, const float* __restrict__ bhh1,
    const float* __restrict__ Wlin, const float* __restrict__ blin,
    _Float16* __restrict__ h0buf, _Float16* __restrict__ h1buf,
    float* __restrict__ c0ws, float* __restrict__ c1ws,
    unsigned* __restrict__ sync, float* __restrict__ out)
{
  __shared__ _Float16 wfrag[NFRAG * 512];   // 104 KB weight image, fragment order
  __shared__ float    redbuf[64];

  const int tid = threadIdx.x;
  const int wg  = blockIdx.x;
  const int hidBase = wg * 16;

  // ---- build f16 weight fragments in LDS (once) ----
  // fragment f: lane L, half n  <->  W[j*256 + hidBase + L%16][kc + k], k = 8(L/16)+n (n<8), 16+8(L/16)+n-8
  for (int idx = tid; idx < NFRAG * 512; idx += 256) {
    int f = idx >> 9;
    int r = idx & 511;
    int L = r >> 4;
    int n = r & 15;
    const float* W; int ld; int lf;
    if (f < 8)       { W = Wih0; ld = 64;  lf = f;      }
    else if (f < 40) { W = Whh0; ld = HID; lf = f - 8;  }
    else if (f < 72) { W = Wih1; ld = HID; lf = f - 40; }
    else             { W = Whh1; ld = HID; lf = f - 72; }
    int kc  = (lf >> 2) * 32;
    int j   = lf & 3;
    int row = j * 256 + hidBase + (L & 15);
    int col = kc + ((n < 8) ? (8 * (L >> 4) + n) : (16 + 8 * (L >> 4) + (n - 8)));
    int off = f * 512 + ((n < 8) ? (L * 8 + n) : (256 + L * 8 + (n - 8)));
    wfrag[off] = (_Float16)W[row * ld + col];
  }
  __syncthreads();

  const int lane  = tid & 31;
  const int wave  = tid >> 5;
  const int m0    = wave * 16;                 // this wave's 16-batch tile
  const int nl    = lane & 15;                 // N column within tile
  const int hidx  = hidBase + nl;              // owned hidden index
  const int koff  = (lane < 16) ? 0 : 8;       // A-fragment K sub-offset
  const int mbase = m0 + ((lane < 16) ? 0 : 8);

  const float bi0 = bih0[hidx]       + bhh0[hidx];
  const float bf0 = bih0[256 + hidx] + bhh0[256 + hidx];
  const float bg0 = bih0[512 + hidx] + bhh0[512 + hidx];
  const float bo0 = bih0[768 + hidx] + bhh0[768 + hidx];
  const float bi1 = bih1[hidx]       + bhh1[hidx];
  const float bf1 = bih1[256 + hidx] + bhh1[256 + hidx];
  const float bg1 = bih1[512 + hidx] + bhh1[512 + hidx];
  const float bo1 = bih1[768 + hidx] + bhh1[768 + hidx];
  const float bl  = blin[0];

  unsigned bar = 0;
  for (int t = 0; t < T_STEPS; ++t) {
    const int cur = t & 1, nxt = cur ^ 1;
    const _Float16* h0c = h0buf + cur * HB;
    _Float16*       h0n = h0buf + nxt * HB;
    const _Float16* h1c = h1buf + cur * HB;
    _Float16*       h1n = h1buf + nxt * HB;
    const float*    xt  = x + (size_t)t * (BATCH * 64);

    // ===== layer 0: z = x_t @ Wih0^T + h0 @ Whh0^T =====
    v8f zi = {}, zf = {}, zg = {}, zo = {};
#pragma unroll
    for (int kcx = 0; kcx < 2; ++kcx) {
      half16 A = load_x_frag(xt, m0 + nl, kcx * 32 + koff);
      wmma4(A, wfrag, kcx * 4, lane, zi, zf, zg, zo);
    }
#pragma unroll
    for (int kch = 0; kch < 8; ++kch) {
      half16 A = load_h_frag(h0c, m0 + nl, kch * 32 + koff);
      wmma4(A, wfrag, 8 + kch * 4, lane, zi, zf, zg, zo);
    }
    gate_update(zi, zf, zg, zo, bi0, bf0, bg0, bo0, c0ws, h0n, mbase, hidx);

    bar += NWG; grid_barrier(sync, bar);

    // ===== layer 1: z = h0_new @ Wih1^T + h1 @ Whh1^T =====
    v8f yi = {}, yf = {}, yg = {}, yo = {};
#pragma unroll
    for (int kch = 0; kch < 8; ++kch) {
      half16 A = load_h_frag(h0n, m0 + nl, kch * 32 + koff);
      wmma4(A, wfrag, 40 + kch * 4, lane, yi, yf, yg, yo);
    }
#pragma unroll
    for (int kch = 0; kch < 8; ++kch) {
      half16 A = load_h_frag(h1c, m0 + nl, kch * 32 + koff);
      wmma4(A, wfrag, 72 + kch * 4, lane, yi, yf, yg, yo);
    }
    gate_update(yi, yf, yg, yo, bi1, bf1, bg1, bo1, c1ws, h1n, mbase, hidx);

    bar += NWG; grid_barrier(sync, bar);

    // ===== head: out[t,b] = h1_new[b,:] . Wlin + b_lin  (8 batches per WG) =====
    if (tid < 64) {
      int bloc = tid >> 3, part = tid & 7;
      int b = wg * 8 + bloc;
      const half8* hp = (const half8*)(h1n + b * HID + part * 32);  // 64B aligned
      const float* wl = Wlin + part * 32;
      float s = 0.f;
#pragma unroll
      for (int c = 0; c < 4; ++c) {
        half8  hv = hp[c];
        float4 w0 = *(const float4*)(wl + c * 8);
        float4 w1 = *(const float4*)(wl + c * 8 + 4);
        s += (float)hv[0] * w0.x + (float)hv[1] * w0.y +
             (float)hv[2] * w0.z + (float)hv[3] * w0.w +
             (float)hv[4] * w1.x + (float)hv[5] * w1.y +
             (float)hv[6] * w1.z + (float)hv[7] * w1.w;
      }
      redbuf[tid] = s;
    }
    __syncthreads();
    if (tid < 8) {
      float s = 0.f;
#pragma unroll
      for (int q = 0; q < 8; ++q) s += redbuf[tid * 8 + q];
      out[(size_t)t * BATCH + wg * 8 + tid] = s + bl;
    }
    // redbuf reuse is protected by the grid barriers' __syncthreads next iter
  }
}

// ---------------- host launch ----------------
extern "C" void kernel_launch(void* const* d_in, const int* in_sizes, int n_in,
                              void* d_out, int out_size, void* d_ws, size_t ws_size,
                              hipStream_t stream) {
  (void)in_sizes; (void)n_in; (void)out_size; (void)ws_size;

  const float* x    = (const float*)d_in[0];
  const float* h0   = (const float*)d_in[1];
  const float* c0   = (const float*)d_in[2];
  const float* Wih0 = (const float*)d_in[3];
  const float* Whh0 = (const float*)d_in[4];
  const float* bih0 = (const float*)d_in[5];
  const float* bhh0 = (const float*)d_in[6];
  const float* Wih1 = (const float*)d_in[7];
  const float* Whh1 = (const float*)d_in[8];
  const float* bih1 = (const float*)d_in[9];
  const float* bhh1 = (const float*)d_in[10];
  const float* Wlin = (const float*)d_in[11];
  const float* blin = (const float*)d_in[12];
  float* out = (float*)d_out;

  char* ws = (char*)d_ws;
  unsigned*  sync  = (unsigned*)(ws + 0);                       // 256 B reserved
  _Float16*  h0buf = (_Float16*)(ws + 256);                     // 2*HB halves
  _Float16*  h1buf = (_Float16*)(ws + 256 + 2 * HB * 2);        // 2*HB halves
  float*     c0ws  = (float*)   (ws + 256 + 4 * HB * 2);        // HB floats
  float*     c1ws  = (float*)   (ws + 256 + 4 * HB * 2 + HB * 4);

  lstm_prep<<<(HB + 255) / 256, 256, 0, stream>>>(h0, c0, h0buf, h1buf, c0ws, c1ws, sync);
  lstm_persistent<<<NWG, 256, 0, stream>>>(x, Wih0, Whh0, bih0, bhh0,
                                           Wih1, Whh1, bih1, bhh1,
                                           Wlin, blin,
                                           h0buf, h1buf, c0ws, c1ws, sync, out);
}
